// hybrid_74474732913283
// MI455X (gfx1250) — matrix operand order repaired and checked by reading, not compiled, not run on hardware
//
#include <hip/hip_runtime.h>
#include <math.h>

typedef __attribute__((ext_vector_type(16))) _Float16 v16h;
typedef __attribute__((ext_vector_type(8)))  float    v8f;

#define B_   16
#define LSEQ 1025
#define TOKN (B_ * LSEQ)

__device__ __forceinline__ float sigm(float x)  { return 1.f / (1.f + __expf(-x)); }
__device__ __forceinline__ float siluf(float x) { return x * sigm(x); }
__device__ __forceinline__ float seluf(float x) {
    const float sc = 1.0507009873554805f, al = 1.6732632423543772f;
    return x > 0.f ? sc * x : sc * al * (__expf(x) - 1.f);
}

// ---------------------------------------------------------------------------
// WMMA GEMM: C[M,N] = act(A[M,K] @ W[K,N] + bias) + resid, M = TOKN.
// block = 128 threads (4 waves); wave -> one 16x16 output tile.
// grid = (TOKN/16, N/64). A tile (16 x K, contiguous rows) is staged to LDS
// via CDNA5 async global->LDS (ASYNCcnt) and reused by all 4 waves; fragments
// follow the CDNA5 16-bit A / B and f32 C VGPR layouts. K compile-time so the
// K-loop fully unrolls into back-to-back v_wmma ops.
// ---------------------------------------------------------------------------
template<int K, bool HAS_BIAS, int ACT, bool HAS_RESID>
__global__ void k_gemm(const float* __restrict__ A, const float* __restrict__ W,
                       const float* __restrict__ bias, const float* __restrict__ resid,
                       float* __restrict__ C, int N)
{
    __shared__ float As[16 * K];

    const int mtile = blockIdx.x;
    const int wave  = threadIdx.x >> 5;
    const int lane  = threadIdx.x & 31;
    const int ntile = blockIdx.y * 4 + wave;

    // ---- async stage of contiguous A tile: 16*K floats, 4 floats/thread/pass
    {
        const float* Atile = A + (size_t)mtile * 16 * K;
        uint32_t lds_base = (uint32_t)(uintptr_t)&As[0];   // LDS aperture: low 32 bits = LDS offset
        constexpr int PASSES = (16 * K) / (128 * 4);
#pragma unroll
        for (int p = 0; p < PASSES; ++p) {
            int f = (p * 128 + (int)threadIdx.x) * 4;
            unsigned long long ga = (unsigned long long)(uintptr_t)(Atile + f);
            uint32_t la = lds_base + (uint32_t)(f * 4);
            asm volatile("global_load_async_to_lds_b128 %0, %1, off"
                         :: "v"(la), "v"(ga) : "memory");
        }
        asm volatile("s_wait_asynccnt 0" ::: "memory");
    }
    __syncthreads();

    if (ntile * 16 < N) {               // wave-uniform guard (EXEC stays all-ones)
        const int half = lane >> 4;
        const int l15  = lane & 15;
        const int coln = ntile * 16 + l15;
        const float2* Arow2 = reinterpret_cast<const float2*>(&As[l15 * K]);

        v8f c = {};
#pragma unroll
        for (int k0 = 0; k0 < K; k0 += 32) {
            v16h a, b;
#pragma unroll
            for (int p = 0; p < 8; ++p) {
                // A 16x32 f16 layout: VGPR p holds K-pair {2p | 16+2(p-4)} + 8*(lane>=16)
                int ka = k0 + ((p < 4) ? (2 * p) : (16 + 2 * (p - 4))) + half * 8;
                float2 av = Arow2[ka >> 1];              // ds_load_b64 from staged tile
                a[2 * p]     = (_Float16)av.x;
                a[2 * p + 1] = (_Float16)av.y;
                // B 32x16 f16 layout: same K-pair order as A: {2p | 16+2(p-4)} + 8*(lane>=16)
                int kb = k0 + ((p < 4) ? (2 * p) : (16 + 2 * (p - 4))) + half * 8;
                b[2 * p]     = (_Float16)W[(size_t)kb * N + coln];
                b[2 * p + 1] = (_Float16)W[(size_t)(kb + 1) * N + coln];
            }
            c = __builtin_amdgcn_wmma_f32_16x16x32_f16(false, a, false, b, (short)0, c,
                                                       false, false);
        }
#pragma unroll
        for (int r = 0; r < 8; ++r) {
            size_t row = (size_t)(mtile * 16 + r + half * 8);
            float v = c[r];
            if (HAS_BIAS)  v += bias[coln];
            if (ACT == 1)  v = seluf(v);
            if (HAS_RESID) v += resid[row * N + coln];
            C[row * N + coln] = v;
        }
    }
}

// ---------------------------------------------------------------------------
// Embedding + CLS, written to both h0 (TCN input) and hm (block stream)
// ---------------------------------------------------------------------------
__global__ void k_embed(const float* __restrict__ x, const float* __restrict__ cls,
                        const float* __restrict__ ew, const float* __restrict__ eb,
                        float* __restrict__ h0, float* __restrict__ hm)
{
    int i = blockIdx.x * blockDim.x + threadIdx.x;
    if (i >= TOKN * 64) return;
    int c = i & 63, tok = i >> 6;
    int t = tok % LSEQ, b = tok / LSEQ;
    float v;
    if (t == 0) {
        v = cls[c];
    } else {
        v = eb[c];
        const float* xr = x + ((size_t)b * 1024 + (t - 1)) * 32;
#pragma unroll
        for (int k = 0; k < 32; ++k) v += xr[k] * ew[k * 64 + c];
    }
    h0[i] = v; hm[i] = v;
}

__global__ void k_mask(const int* __restrict__ len, float* __restrict__ mask)
{
    int i = blockIdx.x * blockDim.x + threadIdx.x;
    if (i >= TOKN) return;
    int t = i % LSEQ, b = i / LSEQ;
    mask[i] = (t < len[b]) ? 1.f : 0.f;
}

// layernorm over 64 channels; optional mask pre-scale; in-place safe
__global__ void k_layernorm(const float* __restrict__ in, const float* __restrict__ g,
                            const float* __restrict__ be, const float* __restrict__ mask,
                            float* __restrict__ out)
{
    int tok = blockIdx.x * blockDim.x + threadIdx.x;
    if (tok >= TOKN) return;
    float mk = mask ? mask[tok] : 1.f;
    const float* r = in + (size_t)tok * 64;
    float s = 0.f, sq = 0.f;
#pragma unroll
    for (int c = 0; c < 64; ++c) { float v = r[c] * mk; s += v; sq += v * v; }
    float mean = s * (1.f / 64.f);
    float var  = sq * (1.f / 64.f) - mean * mean;
    float inv  = rsqrtf(var + 1e-5f);
    float* o = out + (size_t)tok * 64;
#pragma unroll
    for (int c = 0; c < 64; ++c) o[c] = (r[c] * mk - mean) * inv * g[c] + be[c];
}

// depthwise causal conv (K=12) on xi_raw = xz[:, :128], direction-aware, +SiLU
__global__ void k_conv_silu(const float* __restrict__ xz, const float* __restrict__ cw,
                            const float* __restrict__ cb, float* __restrict__ xi, int dir)
{
    int i = blockIdx.x * blockDim.x + threadIdx.x;
    if (i >= TOKN * 128) return;
    int d = i & 127, tok = i >> 7;
    int t = tok % LSEQ, b = tok / LSEQ;
    float acc = 0.f;
#pragma unroll
    for (int k = 0; k < 12; ++k) {
        int tl = t - 11 + k;
        if (tl >= 0) {
            int tn = dir ? (LSEQ - 1 - tl) : tl;
            acc += cw[d * 12 + k] * xz[((size_t)b * LSEQ + tn) * 256 + d];
        }
    }
    acc += cb[d];
    int tn = dir ? (LSEQ - 1 - t) : t;
    xi[((size_t)b * LSEQ + tn) * 128 + d] = siluf(acc);
}

// x_proj: [TOKN,128] @ [128,36] -> xd (dt:0..3, B:4..19, C:20..35)
__global__ void k_xproj(const float* __restrict__ xi, const float* __restrict__ W,
                        float* __restrict__ xd)
{
    int i = blockIdx.x * blockDim.x + threadIdx.x;
    if (i >= TOKN * 36) return;
    int j = i % 36, tok = i / 36;
    const float* r = xi + (size_t)tok * 128;
    float acc = 0.f;
    for (int c = 0; c < 128; ++c) acc += r[c] * W[c * 36 + j];
    xd[i] = acc;
}

// dt = softplus(xd[:, :4] @ dt_w + dt_b)
__global__ void k_dt(const float* __restrict__ xd, const float* __restrict__ W,
                     const float* __restrict__ bia, float* __restrict__ dt)
{
    int i = blockIdx.x * blockDim.x + threadIdx.x;
    if (i >= TOKN * 128) return;
    int d = i & 127, tok = i >> 7;
    float v = bia[d];
#pragma unroll
    for (int r = 0; r < 4; ++r) v += xd[(size_t)tok * 36 + r] * W[r * 128 + d];
    dt[i] = (v > 20.f) ? v : log1pf(__expf(v));
}

// selective scan: one thread per (b,d); 16 states in registers
__global__ void k_scan(const float* __restrict__ xi, const float* __restrict__ dt,
                       const float* __restrict__ xd, const float* __restrict__ Alog,
                       const float* __restrict__ Dp, float* __restrict__ ys,
                       int dir, int accum)
{
    int id = blockIdx.x * blockDim.x + threadIdx.x;
    if (id >= B_ * 128) return;
    int b = id >> 7, d = id & 127;
    float Ac[16], h[16];
#pragma unroll
    for (int n = 0; n < 16; ++n) { Ac[n] = -__expf(Alog[d * 16 + n]); h[n] = 0.f; }
    float Dd = Dp[d];
    for (int t = 0; t < LSEQ; ++t) {
        int tn = dir ? (LSEQ - 1 - t) : t;
        size_t tok = (size_t)b * LSEQ + tn;
        float u  = xi[tok * 128 + d];
        float dv = dt[tok * 128 + d];
        const float* xr = xd + tok * 36;
        float y = 0.f;
#pragma unroll
        for (int n = 0; n < 16; ++n) {
            float dA = __expf(dv * Ac[n]);
            h[n] = dA * h[n] + dv * xr[4 + n] * u;
            y += h[n] * xr[20 + n];
        }
        y += u * Dd;
        if (accum) ys[tok * 128 + d] += y; else ys[tok * 128 + d] = y;
    }
}

// g = (y_f + y_b) * silu(z), z = xz[:, 128:256]
__global__ void k_gate(const float* __restrict__ ys, const float* __restrict__ xz,
                       float* __restrict__ g)
{
    int i = blockIdx.x * blockDim.x + threadIdx.x;
    if (i >= TOKN * 128) return;
    int d = i & 127, tok = i >> 7;
    float z = xz[(size_t)tok * 256 + 128 + d];
    g[i] = ys[i] * siluf(z);
}

// TCN layer: dilated conv3 (symmetric pad) + batchnorm + relu
__global__ void k_tcn(const float* __restrict__ in, const float* __restrict__ w,
                      const float* __restrict__ bia, const float* __restrict__ g,
                      const float* __restrict__ beta, const float* __restrict__ mean,
                      const float* __restrict__ var, float* __restrict__ out, int dil)
{
    int i = blockIdx.x * blockDim.x + threadIdx.x;
    if (i >= TOKN * 64) return;
    int oc = i & 63, tok = i >> 6;
    int t = tok % LSEQ, b = tok / LSEQ;
    float acc = bia[oc];
#pragma unroll
    for (int k = 0; k < 3; ++k) {
        int tt = t + (k - 1) * dil;
        if (tt >= 0 && tt < LSEQ) {
            const float* r = in + ((size_t)b * LSEQ + tt) * 64;
            const float* wr = w + (size_t)oc * 64 * 3 + k;
            for (int ic = 0; ic < 64; ++ic) acc += wr[ic * 3] * r[ic];
        }
    }
    acc = (acc - mean[oc]) * rsqrtf(var[oc] + 1e-5f) * g[oc] + beta[oc];
    out[i] = fmaxf(acc, 0.f);
}

__global__ void k_concat(const float* __restrict__ hm, const float* __restrict__ ht,
                         float* __restrict__ xo)
{
    int i = blockIdx.x * blockDim.x + threadIdx.x;
    if (i >= TOKN * 128) return;
    int c = i & 127, tok = i >> 7;
    xo[i] = (c < 64) ? hm[(size_t)tok * 64 + c] : ht[(size_t)tok * 64 + (c - 64)];
}

// gated attention score per token (masked)
__global__ void k_poolscore(const float* __restrict__ xo,
                            const float* __restrict__ Vw, const float* __restrict__ Vb,
                            const float* __restrict__ Uw, const float* __restrict__ Ub,
                            const float* __restrict__ ww, const float* __restrict__ wb,
                            const float* __restrict__ mask, float* __restrict__ sc)
{
    int tok = blockIdx.x * blockDim.x + threadIdx.x;
    if (tok >= TOKN) return;
    const float* r = xo + (size_t)tok * 128;
    float s = wb[0];
    for (int h = 0; h < 16; ++h) {
        float v = Vb[h], u = Ub[h];
        for (int c = 0; c < 128; ++c) { float xc = r[c]; v += xc * Vw[c * 16 + h]; u += xc * Uw[c * 16 + h]; }
        s += ww[h] * tanhf(v) * sigm(u);
    }
    sc[tok] = (mask[tok] > 0.f) ? s : -1e9f;
}

// per-batch softmax stats (max, denom) via LDS reduction
__global__ void k_poolstats(const float* __restrict__ sc, float* __restrict__ st)
{
    __shared__ float red[256];
    int b = blockIdx.x, tid = threadIdx.x;
    float mx = -1e30f;
    for (int t = tid; t < LSEQ; t += 256) mx = fmaxf(mx, sc[b * LSEQ + t]);
    red[tid] = mx; __syncthreads();
    for (int o = 128; o > 0; o >>= 1) { if (tid < o) red[tid] = fmaxf(red[tid], red[tid + o]); __syncthreads(); }
    mx = red[0]; __syncthreads();
    float sm = 0.f;
    for (int t = tid; t < LSEQ; t += 256) sm += __expf(sc[b * LSEQ + t] - mx);
    red[tid] = sm; __syncthreads();
    for (int o = 128; o > 0; o >>= 1) { if (tid < o) red[tid] += red[tid + o]; __syncthreads(); }
    if (tid == 0) { st[b * 2] = mx; st[b * 2 + 1] = red[0]; }
}

__global__ void k_pooled(const float* __restrict__ sc, const float* __restrict__ st,
                         const float* __restrict__ xo, float* __restrict__ pl)
{
    int i = blockIdx.x * blockDim.x + threadIdx.x;
    if (i >= B_ * 128) return;
    int b = i >> 7, c = i & 127;
    float mx = st[b * 2], inv = 1.f / st[b * 2 + 1];
    float acc = 0.f;
    for (int t = 0; t < LSEQ; ++t)
        acc += __expf(sc[b * LSEQ + t] - mx) * xo[((size_t)b * LSEQ + t) * 128 + c];
    pl[i] = acc * inv;
}

// outputs: x1[16] | x2[16*1024] | x3[16]
__global__ void k_outs(const float* __restrict__ pl, const float* __restrict__ xo,
                       const float* __restrict__ w1, const float* __restrict__ b1,
                       const float* __restrict__ w2, const float* __restrict__ b2,
                       const float* __restrict__ w3, const float* __restrict__ b3,
                       float* __restrict__ out)
{
    int i = blockIdx.x * blockDim.x + threadIdx.x;
    const int N2 = B_ * 1024;
    if (i >= B_ + N2 + B_) return;
    if (i < B_) {
        float acc = b1[0];
        for (int c = 0; c < 128; ++c) acc += pl[(size_t)i * 128 + c] * w1[c];
        out[i] = acc;
    } else if (i < B_ + N2) {
        int j = i - B_, b = j >> 10, t = j & 1023;
        const float* r = xo + ((size_t)b * LSEQ + 1 + t) * 128;
        float acc = b2[0];
        for (int c = 0; c < 128; ++c) acc += r[c] * w2[c];
        out[i] = acc;
    } else {
        int b = i - B_ - N2;
        float acc = b3[0];
        for (int c = 0; c < 128; ++c) acc += pl[(size_t)b * 128 + c] * w3[c];
        out[i] = acc;
    }
}

// ---------------------------------------------------------------------------
static inline int grid1(size_t n) { return (int)((n + 255) / 256); }

extern "C" void kernel_launch(void* const* d_in, const int* in_sizes, int n_in,
                              void* d_out, int out_size, void* d_ws, size_t ws_size,
                              hipStream_t stream)
{
    (void)in_sizes; (void)n_in; (void)out_size; (void)ws_size;
    const float* x    = (const float*)d_in[0];
    const int*   xlen = (const int*)d_in[1];
    auto P = [&](int i) { return (const float*)d_in[i]; };
    // param index map (jax tree-flatten, keys sorted):
    // blk3 base=2, blk4 base=19; within block:
    //  +0 ff_b1 +1 ff_b2 +2 ff_w1 +3 ff_w2 +4 A_log +5 D +6 conv_b +7 conv_w
    //  +8 dt_b +9 dt_w +10 in_proj +11 out_proj +12 x_proj +13 n1_b +14 n1_g +15 n2_b +16 n2_g
    // 36 cls, 37 emb_b, 38 emb_w, 39..44 out{1,2,3}_{b,w},
    // 45 Ub 46 Uw 47 Vb 48 Vw 49 wb 50 ww, tcn[l] base = 51+6l: b,beta,g,mean,var,w

    float* ws = (float*)d_ws;
    size_t off = 0;
    auto alloc = [&](size_t n) { float* p = ws + off; off += (n + 63) & ~((size_t)63); return p; };
    float* h0  = alloc((size_t)TOKN * 64);
    float* msk = alloc(TOKN);
    float* hmA = alloc((size_t)TOKN * 64);
    float* hmB = alloc((size_t)TOKN * 64);
    float* xn  = alloc((size_t)TOKN * 64);
    float* xz  = alloc((size_t)TOKN * 256);   // also reused as ffh after gating
    float* xi  = alloc((size_t)TOKN * 128);   // also reused as g after scans
    float* xd  = alloc((size_t)TOKN * 36);
    float* dtb = alloc((size_t)TOKN * 128);
    float* ys  = alloc((size_t)TOKN * 128);
    float* mb  = alloc((size_t)TOKN * 64);
    float* t0  = alloc((size_t)TOKN * 64);
    float* t1  = alloc((size_t)TOKN * 64);
    float* xo  = alloc((size_t)TOKN * 128);
    float* sc  = alloc(TOKN);
    float* st  = alloc(64);
    float* pl  = alloc((size_t)B_ * 128);
    float* ffh = xz;

    // 1) embedding + CLS + mask
    k_embed<<<grid1((size_t)TOKN * 64), 256, 0, stream>>>(x, P(36), P(38), P(37), h0, hmA);
    k_mask<<<grid1(TOKN), 256, 0, stream>>>(xlen, msk);

    // 2) TCN tower on h0 (independent of blocks)
    {
        const float* tin = h0;
        float* touts[5] = { t0, t1, t0, t1, t0 };
        int dil = 1;
        for (int l = 0; l < 5; ++l) {
            int tb = 51 + 6 * l;
            k_tcn<<<grid1((size_t)TOKN * 64), 256, 0, stream>>>(
                tin, P(tb + 5), P(tb + 0), P(tb + 2), P(tb + 1), P(tb + 3), P(tb + 4),
                touts[l], dil);
            tin = touts[l]; dil <<= 1;
        }
    }

    // 3) two BiMamba blocks
    float* hin = hmA; float* hout = hmB;
    for (int blk = 0; blk < 2; ++blk) {
        const int bb = (blk == 0) ? 2 : 19;
        k_layernorm<<<grid1(TOKN), 256, 0, stream>>>(hin, P(bb + 14), P(bb + 13), nullptr, xn);
        // xz = xn @ in_proj  [16400,64]x[64,256]
        k_gemm<64, false, 0, false><<<dim3(TOKN / 16, 4), 128, 0, stream>>>(
            xn, P(bb + 10), nullptr, nullptr, xz, 256);
        for (int dir = 0; dir < 2; ++dir) {
            k_conv_silu<<<grid1((size_t)TOKN * 128), 256, 0, stream>>>(xz, P(bb + 7), P(bb + 6), xi, dir);
            k_xproj<<<grid1((size_t)TOKN * 36), 256, 0, stream>>>(xi, P(bb + 12), xd);
            k_dt<<<grid1((size_t)TOKN * 128), 256, 0, stream>>>(xd, P(bb + 9), P(bb + 8), dtb);
            k_scan<<<8, 256, 0, stream>>>(xi, dtb, xd, P(bb + 4), P(bb + 5), ys, dir, dir);
        }
        k_gate<<<grid1((size_t)TOKN * 128), 256, 0, stream>>>(ys, xz, xi);
        // m_raw = g @ out_proj  [16400,128]x[128,64]
        k_gemm<128, false, 0, false><<<dim3(TOKN / 16, 1), 128, 0, stream>>>(
            xi, P(bb + 11), nullptr, nullptr, mb, 64);
        // mask then layernorm n2 (in place)
        k_layernorm<<<grid1(TOKN), 256, 0, stream>>>(mb, P(bb + 16), P(bb + 15), msk, mb);
        // ffh = selu(mn @ ff_w1 + b1)  [16400,64]x[64,256]
        k_gemm<64, true, 1, false><<<dim3(TOKN / 16, 4), 128, 0, stream>>>(
            mb, P(bb + 2), P(bb + 0), nullptr, ffh, 256);
        // hout = ffh @ ff_w2 + b2 + hin  [16400,256]x[256,64]
        k_gemm<256, true, 0, true><<<dim3(TOKN / 16, 1), 128, 0, stream>>>(
            ffh, P(bb + 3), P(bb + 1), hin, hout, 64);
        float* tmp = hin; hin = hout; hout = tmp;
    }

    // 4) concat [hm | tcn] and gated pooling
    k_concat<<<grid1((size_t)TOKN * 128), 256, 0, stream>>>(hin, t0, xo);
    k_poolscore<<<grid1(TOKN), 256, 0, stream>>>(xo, P(48), P(47), P(46), P(45), P(50), P(49), msk, sc);
    k_poolstats<<<B_, 256, 0, stream>>>(sc, st);
    k_pooled<<<grid1((size_t)B_ * 128), 256, 0, stream>>>(sc, st, xo, pl);

    // 5) heads -> d_out = [x1 | x2 | x3]
    k_outs<<<grid1((size_t)B_ + B_ * 1024 + B_), 256, 0, stream>>>(
        pl, xo, P(40), P(39), P(42), P(41), P(44), P(43), (float*)d_out);
}
